// EZLSTM_37821482008988
// MI455X (gfx1250) — matrix operand order repaired and checked by reading, not compiled, not run on hardware
//
#include <hip/hip_runtime.h>
#include <hip/hip_bf16.h>

// LSTM: B=64, T=512, D=1024, H=1024.
// bf16 WMMA (v_wmma_f32_16x16x32_bf16), f32 accumulate.
// U/W packed once per launch into WMMA B-fragment layout; h packed into
// A-fragment layout by each step kernel for the next step.
// Step kernel: 8 waves = (gate 0..3) x (k-half 0..1). Fragments are staged
// global -> LDS with async copies (ASYNCcnt) in a double-buffered chunk
// pipeline, WMMAs consume from LDS via ds_load_b128.

typedef __attribute__((ext_vector_type(16))) __bf16         v16bf;
typedef __attribute__((ext_vector_type(16))) unsigned short v16u;
typedef __attribute__((ext_vector_type(8)))  float          v8f;
typedef __attribute__((ext_vector_type(4)))  int            v4i;

#define B_  64
#define T_  512
#define D_  1024
#define H_  1024
#define FRAG_U16 512   // ushorts per 32x16 (or 16x32) bf16 fragment (32 lanes * 16)
#define CH  4          // kt per chunk
#define NCH 4          // chunks per K-half (16 kt)

#if defined(__HIP_DEVICE_COMPILE__) && __has_builtin(__builtin_amdgcn_global_load_async_to_lds_b128)
#define HAS_ASYNC 1
#else
#define HAS_ASYNC 0
#endif

// global (addrspace(1) == HIP "__device__") int4 pointer, non-const
#define GPTR(p) ((__attribute__((address_space(1))) v4i*)(unsigned long long)(const void*)(p))
// LDS (addrspace(3)) int4 pointer: low 32 bits of generic shared pointer
#define LPTR(p) ((__attribute__((address_space(3))) v4i*)(unsigned int)(unsigned long long)(const void*)(p))

static __device__ __forceinline__ unsigned short f2bf(float f) {
    unsigned int u = __float_as_uint(f);
    unsigned int r = u + 0x7fffu + ((u >> 16) & 1u);   // round-to-nearest-even
    return (unsigned short)(r >> 16);
}
static __device__ __forceinline__ float sigm(float x) {
    return 1.0f / (1.0f + __expf(-x));
}

#define WMMA_BF16(A, Bf, C) \
    __builtin_amdgcn_wmma_f32_16x16x32_bf16(false, (A), false, (Bf), (short)0, (C), false, false)

static __device__ __forceinline__ void wait_async_done() {
#if HAS_ASYNC
#if __has_builtin(__builtin_amdgcn_s_wait_asynccnt)
    __builtin_amdgcn_s_wait_asynccnt(0);
#else
    asm volatile("s_wait_asynccnt 0x0" ::: "memory");
#endif
#endif
}

// Issue one chunk of async global->LDS copies.
// A chunk: 16 KB flat, cooperatively loaded by the 4 gate-waves of a k-half
//          (each wave: 8 x 512B flat slices; lane moves 16B per op).
// B chunk: 4 KB flat per wave (own gate/column tile), 8 x 512B slices.
static __device__ __forceinline__ void issue_chunk(const unsigned short* gA, unsigned short* sA,
                                                   const unsigned short* gB, unsigned short* sB,
                                                   int gate, int lane) {
#if HAS_ASYNC
#pragma unroll
    for (int o = 0; o < 8; ++o) {
        int offA = (gate * 8 + o) * 256 + lane * 8;    // ushort units (512B slices)
        __builtin_amdgcn_global_load_async_to_lds_b128(GPTR(gA + offA), LPTR(sA + offA), 0, 0);
        int offB = o * 256 + lane * 8;
        __builtin_amdgcn_global_load_async_to_lds_b128(GPTR(gB + offB), LPTR(sB + offB), 0, 0);
    }
#else
#pragma unroll
    for (int o = 0; o < 8; ++o) {
        int offA = (gate * 8 + o) * 256 + lane * 8;
        *(v16u*)(sA + offA) = *(const v16u*)(gA + offA);
        int offB = o * 256 + lane * 8;
        *(v16u*)(sB + offB) = *(const v16u*)(gB + offB);
    }
#endif
}

// Compute one chunk (CH kt-steps) from LDS fragments.
static __device__ __forceinline__ void compute_chunk(const unsigned short* sA, const unsigned short* sB,
                                                     int lane, v8f& a0, v8f& a1, v8f& a2, v8f& a3) {
#pragma unroll
    for (int kt = 0; kt < CH; ++kt) {
        v16bf bfr = *(const v16bf*)(sB + kt * FRAG_U16 + lane * 16);
        v16bf f0  = *(const v16bf*)(sA + (kt * 4 + 0) * FRAG_U16 + lane * 16);
        v16bf f1  = *(const v16bf*)(sA + (kt * 4 + 1) * FRAG_U16 + lane * 16);
        v16bf f2  = *(const v16bf*)(sA + (kt * 4 + 2) * FRAG_U16 + lane * 16);
        v16bf f3  = *(const v16bf*)(sA + (kt * 4 + 3) * FRAG_U16 + lane * 16);
        a0 = WMMA_BF16(f0, bfr, a0);
        a1 = WMMA_BF16(f1, bfr, a1);
        a2 = WMMA_BF16(f2, bfr, a2);
        a3 = WMMA_BF16(f3, bfr, a3);
    }
}

// ---------------------------------------------------------------------------
// Pack a K x N = 1024x1024 f32 matrix into WMMA B fragments (bf16).
// B layout (16-bit, 32x16): lane l -> column n16*16 + (l&15);
//   kb=(l<16)?0:8; idx i (0..15) -> K = kt*32 + kb + i + (i>=8 ? 8 : 0).
// dst fragment index = n16*32 + kt, 512 ushorts each.
// ---------------------------------------------------------------------------
__global__ void lstm_pack_b(const float* __restrict__ src, unsigned short* __restrict__ dst) {
    int tid  = blockIdx.x * blockDim.x + threadIdx.x;   // 65536 threads total
    int lane = tid & 31;
    int frag = tid >> 5;                                // 0..2047 = n16*32 + kt
    int kt   = frag & 31;
    int n16  = frag >> 5;
    int kb   = (lane < 16) ? 0 : 8;
    const float* s = src + (size_t)(kt * 32 + kb) * H_ + (n16 * 16 + (lane & 15));
    unsigned short* d = dst + (size_t)frag * FRAG_U16 + lane * 16;
#pragma unroll
    for (int i = 0; i < 16; ++i) d[i] = f2bf(s[(size_t)(i + ((i >= 8) ? 8 : 0)) * H_]);
}

// ---------------------------------------------------------------------------
// Pack x[:,0,:] (64 x 1024) into WMMA A fragments (bf16).
// A layout (16-bit, 16x32): lane l -> row M = mt*16 + (l&15); kb=(l<16)?0:8;
//   idx i (0..15) -> K offset kb + i + (i>=8 ? 8 : 0).
// dst fragment index = kt*4 + mt.
// ---------------------------------------------------------------------------
__global__ void lstm_pack_x0(const float* __restrict__ x, unsigned short* __restrict__ dst) {
    int tid  = blockIdx.x * blockDim.x + threadIdx.x;   // 4096 threads total
    int lane = tid & 31;
    int frag = tid >> 5;                                // 0..127 = kt*4 + mt
    int kt = frag >> 2, mt = frag & 3;
    int kb = (lane < 16) ? 0 : 8;
    int m  = mt * 16 + (lane & 15);
    const float* s = x + (size_t)m * (T_ * D_) + kt * 32;   // x[m, 0, :]
    unsigned short* d = dst + (size_t)frag * FRAG_U16 + lane * 16;
#pragma unroll
    for (int i = 0; i < 16; ++i) {
        int K = kb + i + ((i >= 8) ? 8 : 0);
        d[i] = f2bf(s[K]);
    }
}

// ---------------------------------------------------------------------------
// gates0[g][m][j] = (x0 @ W_g)[m][j] + bias_g[j]   (f32, runs once).
// grid = 64 (n16), block = 256 (8 waves: gate x k-half). LDS reduce of halves.
// ---------------------------------------------------------------------------
__global__ void lstm_gemm0(const unsigned short* __restrict__ Apack,
                           const unsigned short* __restrict__ Wpack,
                           const float* __restrict__ b0, const float* __restrict__ b1,
                           const float* __restrict__ b2, const float* __restrict__ b3,
                           float* __restrict__ gates0) {
    __shared__ float lds_p[2][4][B_][16];
    int lane = threadIdx.x & 31;
    int wave = threadIdx.x >> 5;
    int gate = wave & 3;
    int kh   = wave >> 2;          // 0..1, K half
    int n16  = blockIdx.x;         // 0..63
    int kt0  = kh * 16;
    const unsigned short* Bp = Wpack + ((size_t)gate * 2048 + (size_t)n16 * 32 + kt0) * FRAG_U16 + lane * 16;
    const unsigned short* Ap = Apack + (size_t)(kt0 * 4) * FRAG_U16 + lane * 16;
    v8f a0 = {}, a1 = {}, a2 = {}, a3 = {};
#pragma unroll 4
    for (int kt = 0; kt < 16; ++kt) {
        v16bf bfr = *(const v16bf*)(Bp + kt * FRAG_U16);
        v16bf f0  = *(const v16bf*)(Ap + (kt * 4 + 0) * FRAG_U16);
        v16bf f1  = *(const v16bf*)(Ap + (kt * 4 + 1) * FRAG_U16);
        v16bf f2  = *(const v16bf*)(Ap + (kt * 4 + 2) * FRAG_U16);
        v16bf f3  = *(const v16bf*)(Ap + (kt * 4 + 3) * FRAG_U16);
        a0 = WMMA_BF16(f0, bfr, a0);
        a1 = WMMA_BF16(f1, bfr, a1);
        a2 = WMMA_BF16(f2, bfr, a2);
        a3 = WMMA_BF16(f3, bfr, a3);
    }
    int mb = (lane >= 16) ? 8 : 0;
    int nn = lane & 15;
#pragma unroll
    for (int v = 0; v < 8; ++v) {
        lds_p[kh][gate][ 0 + v + mb][nn] = a0[v];
        lds_p[kh][gate][16 + v + mb][nn] = a1[v];
        lds_p[kh][gate][32 + v + mb][nn] = a2[v];
        lds_p[kh][gate][48 + v + mb][nn] = a3[v];
    }
    __syncthreads();
#pragma unroll
    for (int e = 0; e < 16; ++e) {
        int idx  = e * 256 + threadIdx.x;
        int g    = e >> 2;                 // compile-time per unrolled e
        int colL = idx & 15;
        int m    = (idx >> 4) & 63;
        int j    = n16 * 16 + colL;
        const float* bp = (g == 0) ? b0 : (g == 1) ? b1 : (g == 2) ? b2 : b3;
        gates0[(size_t)g * (B_ * H_) + (size_t)m * H_ + j] =
            lds_p[0][g][m][colL] + lds_p[1][g][m][colL] + bp[j];
    }
}

// ---------------------------------------------------------------------------
// One LSTM step. grid = 64 WGs (each owns 16 H-columns), block = 256
// (8 waves: gate x k-half). Async global->LDS chunk pipeline feeding WMMAs.
// WG wg owns columns [wg*16, wg*16+16) => half of A-fragment kt = wg>>1.
// ---------------------------------------------------------------------------
__global__ void lstm_step(const unsigned short* __restrict__ Ain,
                          unsigned short* __restrict__ Aout,
                          const unsigned short* __restrict__ Upack,
                          const float* __restrict__ gates0,
                          float* __restrict__ c,
                          float* __restrict__ out, int t) {
    __shared__ unsigned short ldsA0[2][CH * 4 * FRAG_U16];     // 2 x 16 KB, A chunks for kh==0
    __shared__ unsigned short ldsA1[2][CH * 4 * FRAG_U16];     // 2 x 16 KB, A chunks for kh==1
    __shared__ unsigned short ldsB[2][8][CH * FRAG_U16];       // 2 x 8 x 4 KB, per-wave B chunks
    __shared__ float lds_p[2][4][B_][16];                      // 32 KB partial gate tiles
    __shared__ float lds_h[B_][16];                            // 4 KB

    int lane = threadIdx.x & 31;
    int wave = threadIdx.x >> 5;
    int gate = wave & 3;
    int kh   = wave >> 2;
    int wg   = blockIdx.x;         // 0..63
    int kt0  = kh * 16;

    const unsigned short* gB = Upack + ((size_t)gate * 2048 + (size_t)wg * 32 + kt0) * FRAG_U16;
    const unsigned short* gA = Ain + (size_t)(kt0 * 4) * FRAG_U16;

    unsigned short* sA[2];
    sA[0] = (kh == 0) ? &ldsA0[0][0] : &ldsA1[0][0];
    sA[1] = (kh == 0) ? &ldsA0[1][0] : &ldsA1[1][0];

    v8f a0 = {}, a1 = {}, a2 = {}, a3 = {};

    // prologue: chunk 0
    issue_chunk(gA, sA[0], gB, &ldsB[0][wave][0], gate, lane);
    wait_async_done();
    __syncthreads();

#pragma unroll
    for (int cix = 0; cix < NCH; ++cix) {
        int cur = cix & 1;
        if (cix + 1 < NCH) {
            int nxt = (cix + 1) & 1;
            issue_chunk(gA + (size_t)(cix + 1) * (CH * 4 * FRAG_U16), sA[nxt],
                        gB + (size_t)(cix + 1) * (CH * FRAG_U16), &ldsB[nxt][wave][0],
                        gate, lane);
        }
        compute_chunk(sA[cur], &ldsB[cur][wave][0], lane, a0, a1, a2, a3);
        if (cix + 1 < NCH) {
            wait_async_done();
            __syncthreads();
        }
    }

    // scatter partial gate tiles to LDS (D layout: M = v + 8*(lane>=16), N = lane&15)
    int mb = (lane >= 16) ? 8 : 0;
    int nn = lane & 15;
#pragma unroll
    for (int v = 0; v < 8; ++v) {
        lds_p[kh][gate][ 0 + v + mb][nn] = a0[v];
        lds_p[kh][gate][16 + v + mb][nn] = a1[v];
        lds_p[kh][gate][32 + v + mb][nn] = a2[v];
        lds_p[kh][gate][48 + v + mb][nn] = a3[v];
    }
    __syncthreads();

    // elementwise: 64 rows x 16 cols = 1024 elems, 256 threads x 4
#pragma unroll
    for (int e = 0; e < 4; ++e) {
        int idx  = e * 256 + threadIdx.x;
        int colL = idx & 15;
        int m    = idx >> 4;           // 0..63
        int j    = wg * 16 + colL;
        float zf = gates0[(size_t)(0 * B_ + m) * H_ + j] + lds_p[0][0][m][colL] + lds_p[1][0][m][colL];
        float zi = gates0[(size_t)(1 * B_ + m) * H_ + j] + lds_p[0][1][m][colL] + lds_p[1][1][m][colL];
        float zo = gates0[(size_t)(2 * B_ + m) * H_ + j] + lds_p[0][2][m][colL] + lds_p[1][2][m][colL];
        float zc = gates0[(size_t)(3 * B_ + m) * H_ + j] + lds_p[0][3][m][colL] + lds_p[1][3][m][colL];
        float fg = sigm(zf), ig = sigm(zi), og = sigm(zo), ch = tanhf(zc);
        size_t cidx = (size_t)m * H_ + j;
        float cv = fg * c[cidx] + ig * ch;
        c[cidx] = cv;
        float h = og * tanhf(cv);
        out[(size_t)m * (T_ * H_) + (size_t)t * H_ + j] = h;
        lds_h[m][colL] = h;
    }
    __syncthreads();

    // pack h into A-fragments for next step. kt = wg>>1, half = wg&1.
    // A layout idx i_global = half*8 + i -> local col = kb + i (kb = (lane<16)?0:8).
    if (threadIdx.x < 128) {
        int kt   = wg >> 1;
        int half = wg & 1;
        int mt   = threadIdx.x >> 5;   // 0..3
        int kb   = (lane < 16) ? 0 : 8;
        unsigned short* dst = Aout + (size_t)(kt * 4 + mt) * FRAG_U16 + lane * 16 + half * 8;
        const float* hrow = &lds_h[mt * 16 + (lane & 15)][0];
#pragma unroll
        for (int i = 0; i < 8; ++i) dst[i] = f2bf(hrow[kb + i]);
    }
}

// ---------------------------------------------------------------------------
// Host launch. Inputs: x, wf, wi, wo, wc, uf, ui, uo, uc, bf, bi, bo, bc.
// ---------------------------------------------------------------------------
extern "C" void kernel_launch(void* const* d_in, const int* in_sizes, int n_in,
                              void* d_out, int out_size, void* d_ws, size_t ws_size,
                              hipStream_t stream) {
    const float* x  = (const float*)d_in[0];
    const float* W[4] = { (const float*)d_in[1], (const float*)d_in[2],
                          (const float*)d_in[3], (const float*)d_in[4] };
    const float* U[4] = { (const float*)d_in[5], (const float*)d_in[6],
                          (const float*)d_in[7], (const float*)d_in[8] };
    const float* bias[4] = { (const float*)d_in[9], (const float*)d_in[10],
                             (const float*)d_in[11], (const float*)d_in[12] };
    float* out = (float*)d_out;

    char* ws = (char*)d_ws;
    const size_t PACK_BYTES = (size_t)4 * 64 * 32 * FRAG_U16 * 2;   // 8 MB per matrix set
    unsigned short* Upack  = (unsigned short*)(ws);                  // 8 MB
    unsigned short* Wpack  = (unsigned short*)(ws + PACK_BYTES);     // 8 MB
    float*          gates0 = (float*)(ws + 2 * PACK_BYTES);          // 1 MB
    float*          cbuf   = (float*)(ws + 2 * PACK_BYTES + (1u << 20));            // 256 KB
    unsigned short* Afrag0 = (unsigned short*)(ws + 2 * PACK_BYTES + (1u << 20) + (256u << 10)); // 128 KB
    unsigned short* Afrag1 = Afrag0 + (128u << 10) / 2;              // 128 KB
    unsigned short* X0pack = Afrag1 + (128u << 10) / 2;              // 128 KB

    // 1. pack W and U into bf16 B fragments
    for (int g = 0; g < 4; ++g) {
        lstm_pack_b<<<256, 256, 0, stream>>>(W[g], Wpack + (size_t)g * 2048 * FRAG_U16);
        lstm_pack_b<<<256, 256, 0, stream>>>(U[g], Upack + (size_t)g * 2048 * FRAG_U16);
    }
    // 2. pack x0 into bf16 A fragments
    lstm_pack_x0<<<16, 256, 0, stream>>>(x, X0pack);

    // 3. zero init: h fragments (buffer 0) and c state
    (void)hipMemsetAsync(Afrag0, 0, 128u << 10, stream);
    (void)hipMemsetAsync(cbuf,   0, 256u << 10, stream);

    // 4. gate pre-activations gates0 = x0 @ W + b
    lstm_gemm0<<<64, 256, 0, stream>>>(X0pack, Wpack,
                                       bias[0], bias[1], bias[2], bias[3], gates0);

    // 5. 512 recurrent steps
    for (int t = 0; t < T_; ++t) {
        unsigned short* Ain  = (t & 1) ? Afrag1 : Afrag0;
        unsigned short* Aout = (t & 1) ? Afrag0 : Afrag1;
        lstm_step<<<64, 256, 0, stream>>>(Ain, Aout, Upack, gates0, cbuf, out, t);
    }
}
